// Attention_17798344475248
// MI455X (gfx1250) — matrix operand order repaired and checked by reading, not compiled, not run on hardware
//
#include <hip/hip_runtime.h>
#include <hip/hip_bf16.h>
#include <math.h>

// ---------------------------------------------------------------------------
// Phasor attention on CDNA5 WMMA (v_wmma_f32_16x16x32_f16).
//   encode(act(z)) == z/|z|  =>  real GEMM pairs + rsqrt; atan2 only at end.
// All GEMM operands are laid out so every WMMA fragment loads as exactly
// two global_load_b128 per lane (no sub-dword packing):
//   - A operands: row-major, K contiguous (native A-fragment layout)
//   - B operands: K contiguous per output column (weights pre-transposed,
//     V written transposed by the multi_dense epilogue)
// ---------------------------------------------------------------------------

typedef __attribute__((ext_vector_type(16))) _Float16 v16h;
typedef __attribute__((ext_vector_type(8)))  _Float16 v8h;
typedef __attribute__((ext_vector_type(8)))  float    v8f;

#define PI_F 3.14159265358979323846f

__device__ __forceinline__ v8f wmma16(v16h a, v16h b, v8f c) {
  return __builtin_amdgcn_wmma_f32_16x16x32_f16(false, a, false, b,
                                                (short)0, c, false, false);
}

// A fragment, 16x32 (MxK), row-major source, pre-offset to (row0, k0).
// Lane L: M = L&15, hi = L>>4.  a[0..7] = K hi*8+0..7, a[8..15] = K 16+hi*8+0..7
// -> two contiguous 16-byte loads per lane.
__device__ __forceinline__ v16h load_frag_a(const _Float16* __restrict__ A,
                                            int lda, int lane) {
  const int l = lane & 15, hi = lane >> 4;
  const _Float16* base = A + l * lda + hi * 8;
  v8h p0 = *(const v8h*)(base);
  v8h p1 = *(const v8h*)(base + 16);
  v16h a;
#pragma unroll
  for (int i = 0; i < 8; ++i) { a[i] = p0[i]; a[8 + i] = p1[i]; }
  return a;
}

// B fragment, 32x16 (KxN), with B[k][n] = Mt[n][k] (Mt row-major along K),
// pre-offset to (n0, k0). Lane L: N = L&15, kh = L>>4; b[0..7] = Mt[n][kh*8+0..7], b[8..15] = Mt[n][16+kh*8+0..7]
// -> two contiguous 16-byte loads per lane.
__device__ __forceinline__ v16h load_frag_b_tr(const _Float16* __restrict__ Mt,
                                               int ldm, int lane) {
  const int n = lane & 15, kh = lane >> 4;
  const _Float16* base = Mt + n * ldm + kh * 8;
  v8h p0 = *(const v8h*)(base);
  v8h p1 = *(const v8h*)(base + 16);
  v16h b;
#pragma unroll
  for (int i = 0; i < 8; ++i) { b[i] = p0[i]; b[8 + i] = p1[i]; }
  return b;
}

// ---------------------------------------------------------------------------
// Elementwise kernels
// ---------------------------------------------------------------------------

__global__ __launch_bounds__(256) void encode_kernel(
    const float* __restrict__ x, _Float16* __restrict__ oc,
    _Float16* __restrict__ os, int n) {
  int i = blockIdx.x * 256 + threadIdx.x;
  if (i < n) {
    float s, c;
    __sincosf(PI_F * x[i], &s, &c);
    oc[i] = (_Float16)c;
    os[i] = (_Float16)s;
  }
}

// Convert f32 -> f16 with transpose: per z-matrix, out[n*K + k] = in[k*N + n].
__global__ __launch_bounds__(256) void cvt_tr_kernel(
    const float* __restrict__ w, _Float16* __restrict__ o, int K, int N) {
  const size_t mat = (size_t)blockIdx.z * K * N;
  int i = blockIdx.x * 256 + threadIdx.x;  // i = k*N + n (coalesced read)
  if (i < K * N) {
    int k = i / N, n = i - k * N;
    o[mat + (size_t)n * K + k] = (_Float16)w[mat + i];
  }
}

// ---------------------------------------------------------------------------
// multi_dense: RE = Ac@W + bias, IM = As@W; output normalized phasor pair.
// WT is the transposed weight [N][K]. Output written with generic strides so
// the V projection can be stored transposed ([d][row]) for the PV GEMM.
// Block = 128 threads = 4 waves; wave -> 16 rows x 64 cols. Grid (N/64, M/64).
// ---------------------------------------------------------------------------
__global__ __launch_bounds__(128) void multidense_kernel(
    const _Float16* __restrict__ Ac, const _Float16* __restrict__ As,
    const _Float16* __restrict__ WT, const float* __restrict__ bias,
    _Float16* __restrict__ Oc, _Float16* __restrict__ Os,
    int M, int N, int K, int out_rs, int out_cs) {
  const int lane = threadIdx.x & 31;
  const int wave = threadIdx.x >> 5;
  const int m0 = blockIdx.y * 64 + wave * 16;
  const int n0 = blockIdx.x * 64;

  v8f accR[4] = {}, accI[4] = {};
  for (int k0 = 0; k0 < K; k0 += 32) {
    v16h ac = load_frag_a(Ac + (size_t)m0 * K + k0, K, lane);
    v16h as = load_frag_a(As + (size_t)m0 * K + k0, K, lane);
#pragma unroll
    for (int j = 0; j < 4; ++j) {
      v16h bf = load_frag_b_tr(WT + (size_t)(n0 + j * 16) * K + k0, K, lane);
      accR[j] = wmma16(ac, bf, accR[j]);
      accI[j] = wmma16(as, bf, accI[j]);
    }
  }
  const int n = lane & 15, mh = lane >> 4;
#pragma unroll
  for (int j = 0; j < 4; ++j) {
    const int ncol = n0 + j * 16 + n;
    const float bv = bias[ncol];
#pragma unroll
    for (int v = 0; v < 8; ++v) {
      const int row = m0 + mh * 8 + v;
      float re = accR[j][v] + bv;
      float im = accI[j][v];
      float inv = rsqrtf(re * re + im * im + 1e-20f);
      const size_t idx = (size_t)row * out_rs + (size_t)ncol * out_cs;
      Oc[idx] = (_Float16)(re * inv);
      Os[idx] = (_Float16)(im * inv);
    }
  }
}

// ---------------------------------------------------------------------------
// scores[b,t,T] = (qc@kc^T + qs@ks^T) / D  (Re of phasor inner product).
// K^T operand is K-contiguous per lane already. Grid: (Tkv/64, Tq/64, B).
// ---------------------------------------------------------------------------
__global__ __launch_bounds__(128) void score_kernel(
    const _Float16* __restrict__ qc, const _Float16* __restrict__ qs,
    const _Float16* __restrict__ kc, const _Float16* __restrict__ ks,
    float* __restrict__ scores, int Tq, int Tkv, int D) {
  const int lane = threadIdx.x & 31;
  const int wave = threadIdx.x >> 5;
  const int b = blockIdx.z;
  const int t0 = blockIdx.y * 64 + wave * 16;
  const int T0 = blockIdx.x * 64;
  const _Float16* qcb = qc + (size_t)b * Tq * D;
  const _Float16* qsb = qs + (size_t)b * Tq * D;
  const _Float16* kcb = kc + (size_t)b * Tkv * D;
  const _Float16* ksb = ks + (size_t)b * Tkv * D;

  v8f acc[4] = {};
  for (int k0 = 0; k0 < D; k0 += 32) {
    v16h aqc = load_frag_a(qcb + (size_t)t0 * D + k0, D, lane);
    v16h aqs = load_frag_a(qsb + (size_t)t0 * D + k0, D, lane);
#pragma unroll
    for (int j = 0; j < 4; ++j) {
      v16h bkc = load_frag_b_tr(kcb + (size_t)(T0 + j * 16) * D + k0, D, lane);
      acc[j] = wmma16(aqc, bkc, acc[j]);
      v16h bks = load_frag_b_tr(ksb + (size_t)(T0 + j * 16) * D + k0, D, lane);
      acc[j] = wmma16(aqs, bks, acc[j]);
    }
  }
  const float scale = 1.0f / (float)D;
  const int n = lane & 15, mh = lane >> 4;
#pragma unroll
  for (int j = 0; j < 4; ++j) {
    const int Tcol = T0 + j * 16 + n;
#pragma unroll
    for (int v = 0; v < 8; ++v) {
      const int trow = t0 + mh * 8 + v;
      scores[((size_t)b * Tq + trow) * Tkv + Tcol] = acc[j][v] * scale;
    }
  }
}

// ---------------------------------------------------------------------------
// Row softmax (f32 in, f16 probs out). One wave per row, 8 waves per block.
// ---------------------------------------------------------------------------
__global__ __launch_bounds__(256) void softmax_kernel(
    const float* __restrict__ s, _Float16* __restrict__ p, int cols) {
  const int wave = threadIdx.x >> 5, lane = threadIdx.x & 31;
  const int row = blockIdx.x * 8 + wave;
  const float* sr = s + (size_t)row * cols;
  _Float16* pr = p + (size_t)row * cols;

  float m = -1e30f;
  for (int c = lane; c < cols; c += 32) m = fmaxf(m, sr[c]);
#pragma unroll
  for (int o = 16; o > 0; o >>= 1) m = fmaxf(m, __shfl_xor(m, o, 32));

  float sum = 0.f;
  for (int c = lane; c < cols; c += 32) sum += __expf(sr[c] - m);
#pragma unroll
  for (int o = 16; o > 0; o >>= 1) sum += __shfl_xor(sum, o, 32);

  const float inv = 1.0f / sum;
  for (int c = lane; c < cols; c += 32)
    pr[c] = (_Float16)(__expf(sr[c] - m) * inv);
}

// ---------------------------------------------------------------------------
// PV: OR = P@vc, OI = P@vs with V stored transposed vT[d][b*T+t]; normalize;
// scatter into [b,t,h*D] layout for the final projection. Grid (D/64, Tq/64, B).
// ---------------------------------------------------------------------------
__global__ __launch_bounds__(128) void pv_kernel(
    const _Float16* __restrict__ P, const _Float16* __restrict__ vcT,
    const _Float16* __restrict__ vsT, _Float16* __restrict__ Oc,
    _Float16* __restrict__ Os, int head, int Tq, int D, int Tkv, int HD,
    int Mtot) {
  const int lane = threadIdx.x & 31;
  const int wave = threadIdx.x >> 5;
  const int b = blockIdx.z;
  const int t0 = blockIdx.y * 64 + wave * 16;
  const int n0 = blockIdx.x * 64;
  const _Float16* Pb = P + (size_t)b * Tq * Tkv;

  v8f accR[4] = {}, accI[4] = {};
  for (int k0 = 0; k0 < Tkv; k0 += 32) {
    v16h ap = load_frag_a(Pb + (size_t)t0 * Tkv + k0, Tkv, lane);
#pragma unroll
    for (int j = 0; j < 4; ++j) {
      const size_t boff = (size_t)(n0 + j * 16) * Mtot + (size_t)b * Tkv + k0;
      v16h bc = load_frag_b_tr(vcT + boff, Mtot, lane);
      accR[j] = wmma16(ap, bc, accR[j]);
      v16h bs = load_frag_b_tr(vsT + boff, Mtot, lane);
      accI[j] = wmma16(ap, bs, accI[j]);
    }
  }
  const int n = lane & 15, mh = lane >> 4;
#pragma unroll
  for (int j = 0; j < 4; ++j) {
    const int d = n0 + j * 16 + n;
#pragma unroll
    for (int v = 0; v < 8; ++v) {
      const int t = t0 + mh * 8 + v;
      float re = accR[j][v];
      float im = accI[j][v];
      float inv = rsqrtf(re * re + im * im + 1e-20f);
      const size_t idx = ((size_t)b * Tq + t) * HD + head * D + d;
      Oc[idx] = (_Float16)(re * inv);
      Os[idx] = (_Float16)(im * inv);
    }
  }
}

// ---------------------------------------------------------------------------
// Final projection: ZR = oc@wo + bo, ZI = os@wo; out = atan2(ZI, ZR)/pi (f32).
// WoT is transposed [D][HD].
// ---------------------------------------------------------------------------
__global__ __launch_bounds__(128) void final_kernel(
    const _Float16* __restrict__ Ac, const _Float16* __restrict__ As,
    const _Float16* __restrict__ WoT, const float* __restrict__ bias,
    float* __restrict__ out, int M, int N, int K) {
  const int lane = threadIdx.x & 31;
  const int wave = threadIdx.x >> 5;
  const int m0 = blockIdx.y * 64 + wave * 16;
  const int n0 = blockIdx.x * 64;

  v8f accR[4] = {}, accI[4] = {};
  for (int k0 = 0; k0 < K; k0 += 32) {
    v16h ac = load_frag_a(Ac + (size_t)m0 * K + k0, K, lane);
    v16h as = load_frag_a(As + (size_t)m0 * K + k0, K, lane);
#pragma unroll
    for (int j = 0; j < 4; ++j) {
      v16h bf = load_frag_b_tr(WoT + (size_t)(n0 + j * 16) * K + k0, K, lane);
      accR[j] = wmma16(ac, bf, accR[j]);
      accI[j] = wmma16(as, bf, accI[j]);
    }
  }
  const int n = lane & 15, mh = lane >> 4;
#pragma unroll
  for (int j = 0; j < 4; ++j) {
    const int ncol = n0 + j * 16 + n;
    const float bv = bias[ncol];
#pragma unroll
    for (int v = 0; v < 8; ++v) {
      const int row = m0 + mh * 8 + v;
      float re = accR[j][v] + bv;
      float im = accI[j][v];
      out[(size_t)row * N + ncol] = atan2f(im, re) * (1.0f / PI_F);
    }
  }
}

// ---------------------------------------------------------------------------
// Host-side orchestration
// ---------------------------------------------------------------------------
extern "C" void kernel_launch(void* const* d_in, const int* in_sizes, int n_in,
                              void* d_out, int out_size, void* d_ws,
                              size_t ws_size, hipStream_t stream) {
  (void)in_sizes; (void)n_in; (void)out_size; (void)ws_size;

  constexpr int B = 2, T = 1024, D = 512, H = 8;
  constexpr int M = B * T;      // 2048 flattened (b,t) rows
  constexpr int HD = H * D;     // 4096

  const float* query    = (const float*)d_in[0];
  const float* keyvalue = (const float*)d_in[1];
  const float* wq = (const float*)d_in[2];
  const float* bq = (const float*)d_in[3];
  const float* wk = (const float*)d_in[4];
  const float* bk = (const float*)d_in[5];
  const float* wv = (const float*)d_in[6];
  const float* bv = (const float*)d_in[7];
  const float* wo = (const float*)d_in[8];
  const float* bo = (const float*)d_in[9];
  float* out = (float*)d_out;

  // Workspace carve-out (~80 MB). All sub-buffers 16B+ aligned by size.
  char* w = (char*)d_ws;
  auto alloc_h = [&](size_t n) {
    _Float16* p = (_Float16*)w; w += n * sizeof(_Float16); return p;
  };
  auto alloc_f = [&](size_t n) {
    float* p = (float*)w; w += n * sizeof(float); return p;
  };
  _Float16* encq_c  = alloc_h((size_t)M * D);
  _Float16* encq_s  = alloc_h((size_t)M * D);
  _Float16* enckv_c = alloc_h((size_t)M * D);
  _Float16* enckv_s = alloc_h((size_t)M * D);
  _Float16* wqT = alloc_h((size_t)H * D * D);   // [h][n][k]
  _Float16* wkT = alloc_h((size_t)H * D * D);
  _Float16* wvT = alloc_h((size_t)H * D * D);
  _Float16* woT = alloc_h((size_t)D * HD);      // [n][k]
  _Float16* qc = alloc_h((size_t)M * D);        // [row][d]
  _Float16* qs = alloc_h((size_t)M * D);
  _Float16* kc = alloc_h((size_t)M * D);        // [row][d]
  _Float16* ks = alloc_h((size_t)M * D);
  _Float16* vcT = alloc_h((size_t)D * M);       // transposed: [d][row]
  _Float16* vsT = alloc_h((size_t)D * M);
  float*    scores = alloc_f((size_t)B * T * T);
  _Float16* probs  = alloc_h((size_t)B * T * T);
  _Float16* oc = alloc_h((size_t)M * HD);
  _Float16* os = alloc_h((size_t)M * HD);

  // 1) phasor-encode inputs to (cos, sin) f16
  {
    int n = M * D;
    encode_kernel<<<n / 256, 256, 0, stream>>>(query, encq_c, encq_s, n);
    encode_kernel<<<n / 256, 256, 0, stream>>>(keyvalue, enckv_c, enckv_s, n);
  }
  // 2) weights -> f16, transposed to [N][K] per matrix
  {
    cvt_tr_kernel<<<dim3(D * D / 256, 1, H), 256, 0, stream>>>(wq, wqT, D, D);
    cvt_tr_kernel<<<dim3(D * D / 256, 1, H), 256, 0, stream>>>(wk, wkT, D, D);
    cvt_tr_kernel<<<dim3(D * D / 256, 1, H), 256, 0, stream>>>(wv, wvT, D, D);
    cvt_tr_kernel<<<dim3(HD * D / 256, 1, 1), 256, 0, stream>>>(wo, woT, HD, D);
  }

  // 3) per-head pipeline
  const dim3 gemm_grid(D / 64, M / 64);          // (8, 32)
  const dim3 score_grid(T / 64, T / 64, B);      // (16, 16, 2)
  const dim3 pv_grid(D / 64, T / 64, B);         // (8, 16, 2)
  for (int h = 0; h < H; ++h) {
    const _Float16* wqh = wqT + (size_t)h * D * D;
    const _Float16* wkh = wkT + (size_t)h * D * D;
    const _Float16* wvh = wvT + (size_t)h * D * D;
    multidense_kernel<<<gemm_grid, 128, 0, stream>>>(
        encq_c, encq_s, wqh, bq + h * D, qc, qs, M, D, D, /*rs*/ D, /*cs*/ 1);
    multidense_kernel<<<gemm_grid, 128, 0, stream>>>(
        enckv_c, enckv_s, wkh, bk + h * D, kc, ks, M, D, D, /*rs*/ D, /*cs*/ 1);
    multidense_kernel<<<gemm_grid, 128, 0, stream>>>(
        enckv_c, enckv_s, wvh, bv + h * D, vcT, vsT, M, D, D,
        /*rs*/ 1, /*cs*/ M);  // store V transposed [d][row]

    score_kernel<<<score_grid, 128, 0, stream>>>(qc, qs, kc, ks, scores,
                                                 T, T, D);
    softmax_kernel<<<(B * T) / 8, 256, 0, stream>>>(scores, probs, T);
    pv_kernel<<<pv_grid, 128, 0, stream>>>(probs, vcT, vsT, oc, os, h,
                                           T, D, T, HD, M);
  }

  // 4) final projection + atan2
  final_kernel<<<dim3(D / 64, M / 64), 128, 0, stream>>>(
      oc, os, woT, bo, out, M, D, HD);
}
